// MultiHeadCausalSelfAttention_61830349193310
// MI455X (gfx1250) — matrix operand order repaired and checked by reading, not compiled, not run on hardware
//
#include <hip/hip_runtime.h>

// ---------------------------------------------------------------------------
// MI455X (gfx1250, wave32) multi-head causal self-attention, bf16 WMMA path.
// B=2, T=2048, C=1024, H=16, D=64.
// - all fragments are 2 x b128 loads (layout-chosen)
// - async global->LDS double-buffered pipeline
// - DPP16 butterfly reductions for softmax (no LDS traffic)
// ---------------------------------------------------------------------------

typedef __attribute__((ext_vector_type(16))) __bf16 v16bf;
typedef __attribute__((ext_vector_type(8)))  float  v8f;
typedef __attribute__((ext_vector_type(4)))  unsigned int v4u;
typedef __attribute__((ext_vector_type(4)))  int v4i;

#define HD __device__ __forceinline__

constexpr int EMB   = 1024;
constexpr int NH    = 16;
constexpr int HDIM  = 64;
constexpr int TT    = 2048;
constexpr int BB    = 2;
constexpr int MROWS = BB * TT;   // 4096

union FragAB { v16bf v; unsigned short h[16]; unsigned int u[8]; v4u q[2]; };
union FragC  { v8f   v; float f[8]; };

// float -> bf16, round to nearest even
HD unsigned short f2bf(float f) {
    unsigned u = __float_as_uint(f);
    return (unsigned short)((u + 0x7fffu + ((u >> 16) & 1u)) >> 16);
}

HD v8f wmma_bf16(const FragAB& a, const FragAB& b, v8f c) {
    return __builtin_amdgcn_wmma_f32_16x16x32_bf16(
        false, a.v, false, b.v, (short)0, c, false, false);
}

// ---------------------------------------------------------------------------
// DPP16 butterfly reduction across the 16 lanes owning one row set.
// masks {1,2,7,15} generate the full 4-bit xor space; rows of 16 lanes align
// with DPP16 rows on wave32, so these never cross the khalf groups.
// ---------------------------------------------------------------------------
template <int CTRL>
HD float dppf(float x) {
    return __int_as_float(__builtin_amdgcn_update_dpp(
        0, __float_as_int(x), CTRL, 0xF, 0xF, true));
}
HD float red16_max(float x) {
    x = fmaxf(x, dppf<0xB1>(x));    // quad_perm(1,0,3,2) : xor 1
    x = fmaxf(x, dppf<0x4E>(x));    // quad_perm(2,3,0,1) : xor 2
    x = fmaxf(x, dppf<0x141>(x));   // ROW_HALF_MIRROR    : xor 7
    x = fmaxf(x, dppf<0x140>(x));   // ROW_MIRROR         : xor 15
    return x;
}
HD float red16_sum(float x) {
    x += dppf<0xB1>(x);
    x += dppf<0x4E>(x);
    x += dppf<0x141>(x);
    x += dppf<0x140>(x);
    return x;
}

// ---------------------------------------------------------------------------
// 16-byte global->LDS copy: CDNA5 async-to-LDS when available, else direct.
// ---------------------------------------------------------------------------
#if __has_builtin(__builtin_amdgcn_global_load_async_to_lds_b128) && \
    __has_builtin(__builtin_amdgcn_s_wait_asynccnt)
#define HAVE_ASYNC_LDS 1
#else
#define HAVE_ASYNC_LDS 0
#endif

#if HAVE_ASYNC_LDS
typedef __attribute__((address_space(1))) v4i* glb_v4i;
typedef __attribute__((address_space(3))) v4i* lds_v4i;
HD glb_v4i to_glb(const void* p) {
    return (glb_v4i)(unsigned long long)p;             // generic->global (flat VA)
}
HD lds_v4i to_lds(const void* p) {
    return (lds_v4i)(unsigned)(unsigned long long)p;   // generic LDS addr[31:0]
}
#endif

HD void copy16(const unsigned short* g, unsigned short* l) {
#if HAVE_ASYNC_LDS
    __builtin_amdgcn_global_load_async_to_lds_b128(to_glb(g), to_lds(l), 0, 0);
#else
    *(v4u*)l = *(const v4u*)g;
#endif
}
// wait so that at most `n` async ops remain outstanding (in-order completion)
template <int N>
HD void async_wait() {
#if HAVE_ASYNC_LDS
    __builtin_amdgcn_s_wait_asynccnt(N);
#endif
}

// ---------------------------------------------------------------------------
// Fragment loads (ISA per-lane layout tables; all 2 x b128).
// ---------------------------------------------------------------------------
HD void load_fragA(FragAB& f, const unsigned short* row, int khalf) {
    f.q[0] = *(const v4u*)(row + 8 * khalf);        // K =  8k .. 8k+7
    f.q[1] = *(const v4u*)(row + 16 + 8 * khalf);   // K = 16+8k .. 23+8k
}
HD void load_fragB(FragAB& f, const unsigned short* col, int khalf) {
    f.q[0] = *(const v4u*)(col + 8 * khalf);        // K =  8k .. 8k+7
    f.q[1] = *(const v4u*)(col + 16 + 8 * khalf);   // K = 16+8k .. 23+8k
}

// ---------------------------------------------------------------------------
// f32 -> bf16 elementwise
// ---------------------------------------------------------------------------
__global__ __launch_bounds__(256) void cvt_bf16(const float* __restrict__ in,
                                                unsigned short* __restrict__ out,
                                                int n) {
    int i = blockIdx.x * blockDim.x + threadIdx.x;
    int stride = gridDim.x * blockDim.x;
    for (; i < n; i += stride) out[i] = f2bf(in[i]);
}

// ---------------------------------------------------------------------------
// W [out=N][in=K] f32 -> K-panel bf16:  Wt[((k>>5)*N + n)*32 + (k&31)]
// ---------------------------------------------------------------------------
__global__ __launch_bounds__(256) void transpose_to_panels(const float* __restrict__ W,
                                                           unsigned short* __restrict__ Wt) {
    __shared__ float tile[32][33];
    const int bx = blockIdx.x * 32;   // k block
    const int by = blockIdx.y * 32;   // n block
    const int tx = threadIdx.x;
    const int ty = threadIdx.y;
#pragma unroll
    for (int i = 0; i < 32; i += 8)
        tile[ty + i][tx] = W[(by + ty + i) * EMB + bx + tx];
    __syncthreads();
#pragma unroll
    for (int i = 0; i < 32; i += 8) {
        const int k = bx + ty + i;
        const int n = by + tx;
        Wt[((k >> 5) * EMB + n) * 32 + (k & 31)] = f2bf(tile[tx][ty + i]);
    }
}

// ---------------------------------------------------------------------------
// 128x128-tile bf16 GEMM, double-buffered async pipeline.
// MODE 0: scatter bf16 -> [B,H,T,D]          (K)
// MODE 1: scatter bf16 -> [B,H,D,T]          (V transposed)
// MODE 2: scatter bf16 -> [B,H,T,D], *0.125  (Q pre-scaled by 1/sqrt(D))
// MODE 3: f32 out[m*N+n] = acc + bias[n]     (final projection)
// ---------------------------------------------------------------------------
template <int MODE>
__global__ __launch_bounds__(256) void gemm128(const unsigned short* __restrict__ A,
                                               const unsigned short* __restrict__ Bp,
                                               void* __restrict__ outp,
                                               const float* __restrict__ bias) {
    __shared__ unsigned short As[2][128][40];   // row-major A tiles
    __shared__ unsigned short Bs[2][128][40];   // column-major B tiles

    const int tid   = threadIdx.x;
    const int lane  = tid & 31;
    const int wid   = tid >> 5;
    const int ln    = lane & 15;
    const int khalf = lane >> 4;
    const int wm    = wid & 3;
    const int wn    = wid >> 2;
    const int m0    = blockIdx.y * 128;
    const int n0    = blockIdx.x * 128;

    auto issue = [&](int k0, int buf) {   // 4 x 16B async copies per thread
#pragma unroll
        for (int i = 0; i < 2; ++i) {
            int idx = tid + i * 256;
            int r = idx >> 2, c = (idx & 3) << 3;
            copy16(&A[(m0 + r) * EMB + k0 + c], &As[buf][r][c]);
        }
#pragma unroll
        for (int i = 0; i < 2; ++i) {
            int idx = tid + i * 256;
            int n = idx >> 2, c = (idx & 3) << 3;
            copy16(&Bp[((k0 >> 5) * EMB + n0 + n) * 32 + c], &Bs[buf][n][c]);
        }
    };

    FragC acc[2][4];
#pragma unroll
    for (int mf = 0; mf < 2; ++mf)
#pragma unroll
        for (int nf = 0; nf < 4; ++nf)
#pragma unroll
            for (int i = 0; i < 8; ++i) acc[mf][nf].f[i] = 0.0f;

    issue(0, 0);
    for (int k0 = 0; k0 < EMB; k0 += 32) {
        const int buf = (k0 >> 5) & 1;
        if (k0 + 32 < EMB) {
            issue(k0 + 32, buf ^ 1);   // prefetch next tile (other buffer)
            async_wait<4>();           // current tile's 4 copies retired
        } else {
            async_wait<0>();
        }
        __syncthreads();

        FragAB af[2];
#pragma unroll
        for (int mf = 0; mf < 2; ++mf)
            load_fragA(af[mf], &As[buf][wm * 32 + mf * 16 + ln][0], khalf);
#pragma unroll
        for (int nf = 0; nf < 4; ++nf) {
            FragAB bf;
            load_fragB(bf, &Bs[buf][wn * 64 + nf * 16 + ln][0], khalf);
#pragma unroll
            for (int mf = 0; mf < 2; ++mf)
                acc[mf][nf].v = wmma_bf16(af[mf], bf, acc[mf][nf].v);
        }
        __syncthreads();   // all waves done reading buf before it is reused
    }

#pragma unroll
    for (int mf = 0; mf < 2; ++mf)
#pragma unroll
        for (int nf = 0; nf < 4; ++nf)
#pragma unroll
            for (int v = 0; v < 8; ++v) {
                const int gm = m0 + wm * 32 + mf * 16 + v + 8 * khalf;
                const int gn = n0 + wn * 64 + nf * 16 + ln;
                float val = acc[mf][nf].f[v];
                if (MODE == 3) {
                    ((float*)outp)[gm * EMB + gn] = val + bias[gn];
                } else {
                    if (MODE == 2) val *= 0.125f;   // fold 1/sqrt(D) into Q
                    const int b = gm >> 11, t = gm & (TT - 1);
                    const int h = gn >> 6,  d = gn & (HDIM - 1);
                    int idx;
                    if (MODE == 1) idx = ((b * NH + h) * HDIM + d) * TT + t;
                    else           idx = ((b * NH + h) * TT + t) * HDIM + d;
                    ((unsigned short*)outp)[idx] = f2bf(val);
                }
            }
}

// ---------------------------------------------------------------------------
// Flash-style causal attention, double-buffered K/V streaming.
// grid = B*H*(T/128); block = 256 (8 waves); wave owns 16 query rows.
// Q(scaled),K [B,H,T,D] bf16; Vt [B,H,D,T] bf16; out -> [B,T,C] bf16.
// ---------------------------------------------------------------------------
__global__ __launch_bounds__(256) void attn_kernel(const unsigned short* __restrict__ Qg,
                                                   const unsigned short* __restrict__ Kg,
                                                   const unsigned short* __restrict__ Vtg,
                                                   unsigned short* __restrict__ Og) {
    __shared__ unsigned short Ksr[2][32][72];   // [key][d]  (B cols for S)
    __shared__ unsigned short Vsr[2][64][40];   // [d][key]  (B cols for PV)
    __shared__ unsigned short Ps[8][16][40];    // per-wave P round-trip

    const int tid   = threadIdx.x;
    const int lane  = tid & 31;
    const int wid   = tid >> 5;
    const int ln    = lane & 15;
    const int khalf = lane >> 4;

    const int qt = blockIdx.x & 15;
    const int bh = blockIdx.x >> 4;
    const int b  = bh >> 4;
    const int h  = bh & 15;

    const unsigned short* Qp  = Qg  + bh * TT * HDIM;
    const unsigned short* Kp  = Kg  + bh * TT * HDIM;
    const unsigned short* Vtp = Vtg + bh * HDIM * TT;

    const int qbase = qt * 128;
    const int wq    = qbase + wid * 16;

    auto issue = [&](int kb, int buf) {   // 2 x 16B async copies per thread
        int r = tid >> 3, c = (tid & 7) << 3;
        copy16(&Kp[(kb + r) * HDIM + c], &Ksr[buf][r][c]);
        int r2 = tid >> 2, c2 = (tid & 3) << 3;
        copy16(&Vtp[r2 * TT + kb + c2], &Vsr[buf][r2][c2]);
    };

    // Q fragments (A-layout), two 16B global loads each (already * 1/sqrt(D))
    FragAB qf[2];
#pragma unroll
    for (int kk = 0; kk < 2; ++kk) {
        const unsigned short* row = &Qp[(wq + ln) * HDIM + kk * 32];
        qf[kk].q[0] = *(const v4u*)(row + 8 * khalf);
        qf[kk].q[1] = *(const v4u*)(row + 16 + 8 * khalf);
    }

    float mrow[8], lrow[8];
    FragC of[4];
#pragma unroll
    for (int i = 0; i < 8; ++i) { mrow[i] = -1e30f; lrow[i] = 0.0f; }
#pragma unroll
    for (int nt = 0; nt < 4; ++nt)
#pragma unroll
        for (int i = 0; i < 8; ++i) of[nt].f[i] = 0.0f;

    const int nsteps = (qbase + 128) >> 5;   // causal: keys in [0, qbase+128)
    issue(0, 0);
    for (int s = 0; s < nsteps; ++s) {
        const int kb  = s << 5;
        const int buf = s & 1;
        if (s + 1 < nsteps) {
            issue(kb + 32, buf ^ 1);   // prefetch next K/V tile
            async_wait<2>();
        } else {
            async_wait<0>();
        }
        __syncthreads();

        if (kb <= wq + 15) {   // wave-uniform: EXEC all-ones for WMMA
            // S = Q * K^T
            FragC sf[2];
#pragma unroll
            for (int nt = 0; nt < 2; ++nt) {
#pragma unroll
                for (int i = 0; i < 8; ++i) sf[nt].f[i] = 0.0f;
#pragma unroll
                for (int kk = 0; kk < 2; ++kk) {
                    FragAB kf;
                    load_fragB(kf, &Ksr[buf][nt * 16 + ln][kk * 32], khalf);
                    sf[nt].v = wmma_bf16(qf[kk], kf, sf[nt].v);
                }
            }
            // causal mask (kb <= wq -> every row keeps >= 1 live col)
#pragma unroll
            for (int nt = 0; nt < 2; ++nt)
#pragma unroll
                for (int i = 0; i < 8; ++i) {
                    const int kg = kb + nt * 16 + ln;
                    const int qg = wq + i + 8 * khalf;
                    if (kg > qg) sf[nt].f[i] = -1e30f;
                }
            // online softmax (DPP16 reductions, no LDS)
            float corr[8], rsum[8];
#pragma unroll
            for (int i = 0; i < 8; ++i) {
                const float rmax = red16_max(fmaxf(sf[0].f[i], sf[1].f[i]));
                const float mn   = fmaxf(mrow[i], rmax);
                corr[i] = __expf(mrow[i] - mn);
                mrow[i] = mn;
            }
#pragma unroll
            for (int i = 0; i < 8; ++i) {
                const float p0 = __expf(sf[0].f[i] - mrow[i]);
                const float p1 = __expf(sf[1].f[i] - mrow[i]);
                sf[0].f[i] = p0;
                sf[1].f[i] = p1;
                rsum[i] = red16_sum(p0 + p1);
            }
#pragma unroll
            for (int i = 0; i < 8; ++i) {
                lrow[i] = lrow[i] * corr[i] + rsum[i];
#pragma unroll
                for (int nt = 0; nt < 4; ++nt) of[nt].f[i] *= corr[i];
            }
            // C-layout -> A-layout via per-wave LDS (DS ops in-order per wave)
#pragma unroll
            for (int nt = 0; nt < 2; ++nt)
#pragma unroll
                for (int i = 0; i < 8; ++i)
                    Ps[wid][i + 8 * khalf][nt * 16 + ln] = f2bf(sf[nt].f[i]);
            __builtin_amdgcn_wave_barrier();
            FragAB pf;
            load_fragA(pf, &Ps[wid][ln][0], khalf);
            // O += P * V
#pragma unroll
            for (int nt = 0; nt < 4; ++nt) {
                FragAB vf;
                load_fragB(vf, &Vsr[buf][nt * 16 + ln][0], khalf);
                of[nt].v = wmma_bf16(pf, vf, of[nt].v);
            }
        }
        __syncthreads();   // all waves done with buf before it is reused
    }

    // normalize + store to [B,T,C] bf16 (head h -> cols h*64..h*64+63)
#pragma unroll
    for (int i = 0; i < 8; ++i) {
        const float inv = 1.0f / lrow[i];
        const int q = wq + i + 8 * khalf;
        const int rowbase = (b * TT + q) * EMB + h * HDIM;
#pragma unroll
        for (int nt = 0; nt < 4; ++nt)
            Og[rowbase + nt * 16 + ln] = f2bf(of[nt].f[i] * inv);
    }
}

// ---------------------------------------------------------------------------
// Launch
// ---------------------------------------------------------------------------
extern "C" void kernel_launch(void* const* d_in, const int* in_sizes, int n_in,
                              void* d_out, int out_size, void* d_ws, size_t ws_size,
                              hipStream_t stream) {
    (void)in_sizes; (void)n_in; (void)out_size; (void)ws_size;

    const float* x  = (const float*)d_in[0];
    const float* Wq = (const float*)d_in[1];
    const float* Wk = (const float*)d_in[2];
    const float* Wv = (const float*)d_in[3];
    const float* Wo = (const float*)d_in[4];
    const float* bo = (const float*)d_in[5];

    char* ws = (char*)d_ws;                                  // 48 MB used
    unsigned short* xb  = (unsigned short*)(ws);             // 8 MB
    unsigned short* Wtq = (unsigned short*)(ws + (8ull  << 20));
    unsigned short* Wtk = (unsigned short*)(ws + (10ull << 20));
    unsigned short* Wtv = (unsigned short*)(ws + (12ull << 20));
    unsigned short* Wto = (unsigned short*)(ws + (14ull << 20));
    unsigned short* Qg  = (unsigned short*)(ws + (16ull << 20));
    unsigned short* Kg  = (unsigned short*)(ws + (24ull << 20));
    unsigned short* Vtg = (unsigned short*)(ws + (32ull << 20));
    unsigned short* Ao  = (unsigned short*)(ws + (40ull << 20));

    cvt_bf16<<<2048, 256, 0, stream>>>(x, xb, MROWS * EMB);

    dim3 tb(32, 8), tg(32, 32);
    transpose_to_panels<<<tg, tb, 0, stream>>>(Wq, Wtq);
    transpose_to_panels<<<tg, tb, 0, stream>>>(Wk, Wtk);
    transpose_to_panels<<<tg, tb, 0, stream>>>(Wv, Wtv);
    transpose_to_panels<<<tg, tb, 0, stream>>>(Wo, Wto);

    dim3 gg(EMB / 128, MROWS / 128);   // (8, 32)
    gemm128<2><<<gg, 256, 0, stream>>>(xb, Wtq, Qg, nullptr);   // Q (pre-scaled)
    gemm128<0><<<gg, 256, 0, stream>>>(xb, Wtk, Kg, nullptr);   // K
    gemm128<1><<<gg, 256, 0, stream>>>(xb, Wtv, Vtg, nullptr);  // V^T

    attn_kernel<<<BB * NH * (TT / 128), 256, 0, stream>>>(Qg, Kg, Vtg, Ao);

    gemm128<3><<<gg, 256, 0, stream>>>(Ao, Wto, d_out, bo);
}
